// RPPO_53523882443093
// MI455X (gfx1250) — matrix operand order repaired and checked by reading, not compiled, not run on hardware
//
#include <hip/hip_runtime.h>

typedef __bf16 bf16;
typedef __attribute__((ext_vector_type(16))) __bf16 v16bf;
typedef __attribute__((ext_vector_type(8)))  __bf16 v8bf;
typedef __attribute__((ext_vector_type(8)))  float  v8f;

#define T_STEPS 128
#define NBATCH  2048
#define HDIM    128
#define DDIM    128
#define EPS     1e-5f
#define PX      136      // padded pitch (bf16 elems) for sX/sHb: 68 words -> bank rotate 4/row
#define PH      132      // padded pitch (f32 elems) for sH

// ---------------- workspace layout (bytes) ----------------
#define WS_WC  0         // Wc  f32 128x128   (65536)
#define WS_BC  65536     // bc  f32 128       (512)
#define WS_BG  66048     // bg  f32 384       (1536)
#define WS_WG  67584     // Wg  bf16 384x128  (98304)
#define WS_WH  165888    // Whh bf16 384x128  (98304)  -> total 264192 bytes

// ---- fast transcendentals (hardware TRANS32 ops; guarded fallbacks) ----
__device__ __forceinline__ float fast_tanh(float v) {
#if __has_builtin(__builtin_amdgcn_tanhf)
    return __builtin_amdgcn_tanhf(v);
#else
    return 2.f * __builtin_amdgcn_rcpf(1.f + __expf(-2.f * v)) - 1.f;
#endif
}
__device__ __forceinline__ float fast_sig(float v) {
#if __has_builtin(__builtin_amdgcn_tanhf)
    return fmaf(0.5f, __builtin_amdgcn_tanhf(0.5f * v), 0.5f);
#else
    return __builtin_amdgcn_rcpf(1.f + __expf(-v));
#endif
}
__device__ __forceinline__ float fast_rsqrt(float v) {
#if __has_builtin(__builtin_amdgcn_rsqf)
    return __builtin_amdgcn_rsqf(v);
#else
    return rsqrtf(v);
#endif
}

// ---- prep 1: Wc = W2 @ W1 ; bc = W2 @ b1 + b2 ----
__global__ void prep_combine1(const float* __restrict__ W1, const float* __restrict__ b1,
                              const float* __restrict__ W2, const float* __restrict__ b2,
                              float* __restrict__ Wc, float* __restrict__ bc) {
    int i = blockIdx.x, j = threadIdx.x;           // 128 x 128
    float s = 0.f;
    for (int k = 0; k < DDIM; ++k) s += W2[i*DDIM + k] * W1[k*DDIM + j];
    Wc[i*DDIM + j] = s;
    if (j == 0) {
        float t = b2[i];
        for (int k = 0; k < DDIM; ++k) t += W2[i*DDIM + k] * b1[k];
        bc[i] = t;
    }
}

// ---- prep 2: Wg = W_ih @ Wc (bf16) ; bg = W_ih @ bc + b_ih ; Whh -> bf16 ----
__global__ void prep_combine2(const float* __restrict__ W_ih, const float* __restrict__ b_ih,
                              const float* __restrict__ Wc,   const float* __restrict__ bc,
                              const float* __restrict__ W_hh,
                              bf16* __restrict__ Wg, float* __restrict__ bg,
                              bf16* __restrict__ Whh) {
    int i = blockIdx.x, j = threadIdx.x;           // 384 x 128
    float s = 0.f;
    for (int k = 0; k < DDIM; ++k) s += W_ih[i*DDIM + k] * Wc[k*DDIM + j];
    Wg[i*DDIM + j]  = (bf16)s;
    Whh[i*DDIM + j] = (bf16)W_hh[i*DDIM + j];
    if (j == 0) {
        float t = b_ih[i];
        for (int k = 0; k < DDIM; ++k) t += W_ih[i*DDIM + k] * bc[k];
        bg[i] = t;
    }
}

// ---- fused persistent kernel: LN1 + gi-GEMM + GRU scan + LN2, 16 sequences/block ----
__global__ __launch_bounds__(256, 1)
void rppo_fused(const float* __restrict__ x,    const float* __restrict__ hxs,
                const float* __restrict__ masks,
                const float* __restrict__ ln1_w, const float* __restrict__ ln1_b,
                const float* __restrict__ ln2_w, const float* __restrict__ ln2_b,
                const bf16* __restrict__ Wg,     const float* __restrict__ bg,
                const bf16* __restrict__ Whh,    const float* __restrict__ bhh,
                float* __restrict__ out) {
    __shared__ float sH [16 * PH];      // h, f32 (padded)
    __shared__ bf16  sHb[16 * PX];      // masked h, bf16 (padded)
    __shared__ bf16  sX [16 * PX];      // LN1(x), bf16 (padded)
    __shared__ float sL1w[128], sL1b[128], sL2w[128], sL2b[128];

    const int tid  = threadIdx.x;
    const int lane = tid & 31;
    const int wv   = tid >> 5;                // wave id 0..7
    const int n0   = blockIdx.x * 16;         // first batch row of this block

    if (tid < 128) {
        sL1w[tid] = ln1_w[tid]; sL1b[tid] = ln1_b[tid];
        sL2w[tid] = ln2_w[tid]; sL2b[tid] = ln2_b[tid];
    }

    const int row16 = tid >> 4;               // 0..15 : batch row handled in LN phases
    const int cgrp  = (tid & 15) * 8;         // 8-column group in LN phases
    const int arow  = lane & 15;              // WMMA A-matrix row (M)
    const int lh    = lane >> 4;              // lane half (K-half select)
    const int c16   = lane & 15;

    // h0
    #pragma unroll
    for (int j = 0; j < 8; ++j)
        sH[row16*PH + cgrp + j] = hxs[(size_t)(n0 + row16)*HDIM + cgrp + j];

    union AB { v16bf v; v8bf h[2]; };

    // ---- preload ALL weight B-fragments into registers (step-invariant) ----
    // wave wv owns output tiles {wv, wv+8, wv+16} -> gate columns [16wv,16wv+16) of r,z,n
    v16bf bW[12], bH[12];                     // [j*4+kb]
    #pragma unroll
    for (int j = 0; j < 3; ++j) {
        const int nrow = (wv + 8*j)*16 + c16; // output column == weight row
        #pragma unroll
        for (int kb = 0; kb < 4; ++kb) {
            AB u;
            const bf16* pg = Wg  + nrow*DDIM + kb*32 + 16*lh;
            u.h[0] = *(const v8bf*)pg;  u.h[1] = *(const v8bf*)(pg + 8);
            bW[j*4+kb] = u.v;
            const bf16* ph = Whh + nrow*DDIM + kb*32 + 16*lh;
            u.h[0] = *(const v8bf*)ph;  u.h[1] = *(const v8bf*)(ph + 8);
            bH[j*4+kb] = u.v;
        }
    }

    // ---- loop-invariant gate biases, folded out of the accumulators ----
    const int colR = wv*16 + c16;             // r-gate column (also hidden column owned)
    const float bR  = bg[colR]       + bhh[colR];
    const float bZ  = bg[colR + 128] + bhh[colR + 128];
    const float bgN = bg[colR + 256];
    const float bhN = bhh[colR + 256];

    // ---- software-pipelined x / mask prefetch ----
    const float* xptr = x + (size_t)(n0 + row16)*DDIM + cgrp;
    const float* mptr = masks + (n0 + row16);
    float4 pxa = ((const float4*)xptr)[0];
    float4 pxb = ((const float4*)xptr)[1];
    float  pm  = *mptr;

    for (int t = 0; t < T_STEPS; ++t) {
        __syncthreads();   // previous step's readers of sH/sX are done
        const size_t g = (size_t)t * NBATCH + n0 + row16;

        const float4 cxa = pxa, cxb = pxb;
        const float  cm  = pm;
        if (t + 1 < T_STEPS) {                // prefetch next step's x row + mask
            const float* nx = xptr + (size_t)(t + 1) * NBATCH * DDIM;
            pxa = ((const float4*)nx)[0];
            pxb = ((const float4*)nx)[1];
            pm  = mptr[(size_t)(t + 1) * NBATCH];
        }

        // ---------- phase A: LN1(x_row) -> sX ; mask h -> sH,sHb ----------
        float xv[8];
        xv[0]=cxa.x; xv[1]=cxa.y; xv[2]=cxa.z; xv[3]=cxa.w;
        xv[4]=cxb.x; xv[5]=cxb.y; xv[6]=cxb.z; xv[7]=cxb.w;
        float s = 0.f, sq = 0.f;
        #pragma unroll
        for (int j = 0; j < 8; ++j) { s += xv[j]; sq += xv[j]*xv[j]; }
        #pragma unroll
        for (int m = 8; m >= 1; m >>= 1) { s += __shfl_xor(s, m, 16); sq += __shfl_xor(sq, m, 16); }
        float mean = s * (1.f/128.f);
        float rs   = fast_rsqrt(sq * (1.f/128.f) - mean*mean + EPS);
        v8bf xo;
        #pragma unroll
        for (int j = 0; j < 8; ++j)
            xo[j] = (bf16)((xv[j] - mean) * rs * sL1w[cgrp + j] + sL1b[cgrp + j]);
        *(v8bf*)(sX + row16*PX + cgrp) = xo;

        v8bf ho8;
        #pragma unroll
        for (int j = 0; j < 8; ++j) {
            float hv = sH[row16*PH + cgrp + j] * cm;
            sH[row16*PH + cgrp + j] = hv;
            ho8[j] = (bf16)hv;
        }
        *(v8bf*)(sHb + row16*PX + cgrp) = ho8;
        __syncthreads();

        // ---------- phase B: WMMA (inline-0 C start; biases applied in gate phase) ----------
        v8f gi[3] = {v8f{}, v8f{}, v8f{}};
        v8f gh[3] = {v8f{}, v8f{}, v8f{}};
        #pragma unroll
        for (int kb = 0; kb < 4; ++kb) {
            AB ax, ah;
            const bf16* px = sX  + arow*PX + kb*32 + 16*lh;
            ax.h[0] = *(const v8bf*)px;
            ax.h[1] = *(const v8bf*)(px + 8);
            const bf16* ph = sHb + arow*PX + kb*32 + 16*lh;
            ah.h[0] = *(const v8bf*)ph;
            ah.h[1] = *(const v8bf*)(ph + 8);
            #pragma unroll
            for (int j = 0; j < 3; ++j) {
                gi[j] = __builtin_amdgcn_wmma_f32_16x16x32_bf16(
                            false, ax.v, false, bW[j*4+kb], (short)0, gi[j], false, false);
                gh[j] = __builtin_amdgcn_wmma_f32_16x16x32_bf16(
                            false, ah.v, false, bH[j*4+kb], (short)0, gh[j], false, false);
            }
        }

        // ---------- phase C: gates in accumulator layout -> h_new (own cols) ----------
        #pragma unroll
        for (int r = 0; r < 8; ++r) {
            int m = r + 8*lh;                 // C-layout row
            float rr = fast_sig(gi[0][r] + gh[0][r] + bR);
            float zz = fast_sig(gi[1][r] + gh[1][r] + bZ);
            float nn = fast_tanh(gi[2][r] + bgN + rr * (gh[2][r] + bhN));
            float ho = sH[m*PH + colR];       // masked h
            sH[m*PH + colR] = (1.f - zz)*nn + zz*ho;
        }
        __syncthreads();

        // ---------- phase D: LN2(h_new) -> out ----------
        float hv[8]; float s2 = 0.f, q2 = 0.f;
        #pragma unroll
        for (int j = 0; j < 8; ++j) {
            float v = sH[row16*PH + cgrp + j];
            hv[j] = v; s2 += v; q2 += v*v;
        }
        #pragma unroll
        for (int m = 8; m >= 1; m >>= 1) { s2 += __shfl_xor(s2, m, 16); q2 += __shfl_xor(q2, m, 16); }
        float mean2 = s2 * (1.f/128.f);
        float rs2   = fast_rsqrt(q2 * (1.f/128.f) - mean2*mean2 + EPS);
        float oo[8];
        #pragma unroll
        for (int j = 0; j < 8; ++j)
            oo[j] = (hv[j] - mean2) * rs2 * sL2w[cgrp + j] + sL2b[cgrp + j];
        float4* op = (float4*)(out + g*HDIM + cgrp);
        op[0] = make_float4(oo[0], oo[1], oo[2], oo[3]);
        op[1] = make_float4(oo[4], oo[5], oo[6], oo[7]);
    }

    // ---------- hT (raw final h, not layer-normed) ----------
    {
        const size_t base = (size_t)T_STEPS * NBATCH * HDIM;
        #pragma unroll
        for (int j = 0; j < 8; ++j)
            out[base + (size_t)(n0 + row16)*HDIM + cgrp + j] = sH[row16*PH + cgrp + j];
    }
}

extern "C" void kernel_launch(void* const* d_in, const int* in_sizes, int n_in,
                              void* d_out, int out_size, void* d_ws, size_t ws_size,
                              hipStream_t stream) {
    const float* x    = (const float*)d_in[0];
    const float* hxs  = (const float*)d_in[1];
    const float* mk   = (const float*)d_in[2];
    const float* l1w  = (const float*)d_in[3];
    const float* l1b  = (const float*)d_in[4];
    const float* W1   = (const float*)d_in[5];
    const float* b1   = (const float*)d_in[6];
    const float* W2   = (const float*)d_in[7];
    const float* b2   = (const float*)d_in[8];
    const float* Wih  = (const float*)d_in[9];
    const float* bih  = (const float*)d_in[10];
    const float* Whh  = (const float*)d_in[11];
    const float* bhh  = (const float*)d_in[12];
    const float* l2w  = (const float*)d_in[13];
    const float* l2b  = (const float*)d_in[14];
    float* out = (float*)d_out;

    char* ws = (char*)d_ws;                       // needs ~264 KB
    float* Wc   = (float*)(ws + WS_WC);
    float* bc   = (float*)(ws + WS_BC);
    float* bgp  = (float*)(ws + WS_BG);
    bf16*  WgB  = (bf16*) (ws + WS_WG);
    bf16*  WhhB = (bf16*) (ws + WS_WH);

    prep_combine1<<<128, 128, 0, stream>>>(W1, b1, W2, b2, Wc, bc);
    prep_combine2<<<384, 128, 0, stream>>>(Wih, bih, Wc, bc, Whh, WgB, bgp, WhhB);

    rppo_fused<<<128, 256, 0, stream>>>(x, hxs, mk, l1w, l1b, l2w, l2b,
                                        WgB, bgp, WhhB, bhh, out);
}
